// Attention_24678882083285
// MI455X (gfx1250) — matrix operand order repaired and checked by reading, not compiled, not run on hardware
//
#include <hip/hip_runtime.h>
#include <math.h>

// ---------------------------------------------------------------------------
// Types
// ---------------------------------------------------------------------------
typedef __attribute__((ext_vector_type(16))) __bf16 v16bf;
typedef __attribute__((ext_vector_type(8)))  __bf16 v8bf;
typedef __attribute__((ext_vector_type(8)))  float  v8f;

#define N_HEADS    32
#define N_KV_HEADS 8
#define HD         128
#define BATCH      2
#define SEQ        2048
#define DMODEL     4096
#define KVD        1024

// ---------------------------------------------------------------------------
// CDNA5 async global->LDS copy (ASYNCcnt-tracked), with safe fallback
// ---------------------------------------------------------------------------
#if __has_builtin(__builtin_amdgcn_global_load_async_to_lds_b128)
#define HAVE_ASYNC_LDS 1
typedef int v4i_ __attribute__((vector_size(16)));
typedef __attribute__((address_space(1))) v4i_* as1_v4i;
typedef __attribute__((address_space(3))) v4i_* as3_v4i;
__device__ inline void cp16_g2l(const __bf16* g, __bf16* l) {
  __builtin_amdgcn_global_load_async_to_lds_b128((as1_v4i)g, (as3_v4i)l, 0, 0);
}
__device__ inline void wait_async() {
#if __has_builtin(__builtin_amdgcn_s_wait_asynccnt)
  __builtin_amdgcn_s_wait_asynccnt(0);
#else
  asm volatile("s_wait_asynccnt 0" ::: "memory");
#endif
}
#else
#define HAVE_ASYNC_LDS 0
__device__ inline void cp16_g2l(const __bf16* g, __bf16* l) {
  *(v8bf*)l = *(const v8bf*)g;   // sync: global load + ds store
}
__device__ inline void wait_async() {}
#endif

// ---------------------------------------------------------------------------
// WMMA helpers (CDNA5 bf16 16x16x32, f32 accumulate)
// ---------------------------------------------------------------------------
__device__ inline v8f wmma_bf16(v16bf a, v16bf b, v8f c) {
  return __builtin_amdgcn_wmma_f32_16x16x32_bf16(
      /*neg_a=*/false, a, /*neg_b=*/false, b,
      /*c_mod=*/(short)0, c, /*reuse_a=*/false, /*reuse_b=*/false);
}

// A-matrix fragment (16x32 bf16): lane holds row (lane&15);
// elements 0..7 = K{8h..8h+7}, elements 8..15 = K{16+8h..16+8h+7}
__device__ inline v16bf load_afrag(const __bf16* row, int h) {
  v8bf lo = *(const v8bf*)(row + 8 * h);
  v8bf hi = *(const v8bf*)(row + 16 + 8 * h);
  v16bf o;
#pragma unroll
  for (int i = 0; i < 8; i++) { o[i] = lo[i]; o[i + 8] = hi[i]; }
  return o;
}

// B-matrix fragment (32x16 bf16 = K x N): lane holds column (lane&15);
// elements 0..7 = K{8h..8h+7}, elements 8..15 = K{16+8h..16+8h+7}, read from row n of B^T storage.
__device__ inline v16bf load_bfrag(const __bf16* row, int h) {
  v8bf lo = *(const v8bf*)(row + 8 * h);
  v8bf hi = *(const v8bf*)(row + 16 + 8 * h);
  v16bf o;
#pragma unroll
  for (int i = 0; i < 8; i++) { o[i] = lo[i]; o[i + 8] = hi[i]; }
  return o;
}

// ---------------------------------------------------------------------------
// fp32 -> bf16 conversion
// ---------------------------------------------------------------------------
__global__ void cvt_f32_bf16(const float* __restrict__ in, __bf16* __restrict__ out,
                             size_t n) {
  size_t i = (size_t)blockIdx.x * 256u + threadIdx.x;
  if (i < n) out[i] = (__bf16)in[i];
}

// ---------------------------------------------------------------------------
// NT GEMM: C[M,N] (f32) = A[M,K] (bf16, row-major) * B[N,K]^T (bf16, row-major)
// Block tile 128x64, K-step 64, double-buffered LDS, 256 threads = 8 waves
// in a 4(M) x 2(N) grid; each wave owns a 32x32 output (2x2 WMMA, frag reuse).
// ---------------------------------------------------------------------------
#define BM   128
#define BN   64
#define BK   64
#define KSTR 72   // padded bf16 row stride (144B) to spread LDS banks

__global__ __launch_bounds__(256) void gemm_nt_bf16(
    const __bf16* __restrict__ A, const __bf16* __restrict__ B,
    float* __restrict__ C, int M, int N, int K) {
  __shared__ __align__(16) __bf16 As[2][BM * KSTR];
  __shared__ __align__(16) __bf16 Bs[2][BN * KSTR];

  const int tid  = threadIdx.x;
  const int lane = tid & 31;
  const int w    = tid >> 5;
  const int wm   = w >> 1;  // 4 M-subtiles of 32
  const int wn   = w & 1;   // 2 N-subtiles of 32
  const int m0   = blockIdx.y * BM;
  const int n0   = blockIdx.x * BN;
  const int r    = lane & 15;
  const int hlf  = lane >> 4;

  // staging coords: A 128x64 (2 thr/row, 32 cols each); B 64x64 (4 thr/row)
  const int ar = tid >> 1, ac = (tid & 1) * 32;
  const int br = tid >> 2, bc = (tid & 3) * 16;

  v8f acc[2][2] = {};

  auto stage = [&](int buf, int k0) {
    const __bf16* ga = &A[(size_t)(m0 + ar) * K + k0 + ac];
    __bf16* la = &As[buf][ar * KSTR + ac];
#pragma unroll
    for (int j = 0; j < 4; j++) cp16_g2l(ga + j * 8, la + j * 8);
    const __bf16* gb = &B[(size_t)(n0 + br) * K + k0 + bc];
    __bf16* lb = &Bs[buf][br * KSTR + bc];
    cp16_g2l(gb, lb);
    cp16_g2l(gb + 8, lb + 8);
  };

  const int nkt = K / BK;
  stage(0, 0);
  int cur = 0;
  for (int kt = 0; kt < nkt; kt++) {
    wait_async();
    __syncthreads();
    if (kt + 1 < nkt) stage(cur ^ 1, (kt + 1) * BK);

    const __bf16* Ab = &As[cur][0];
    const __bf16* Bb = &Bs[cur][0];
#pragma unroll
    for (int s = 0; s < 2; s++) {  // two K=32 sub-steps
      v16bf a0 = load_afrag(Ab + (wm * 32 + r) * KSTR + 32 * s, hlf);
      v16bf a1 = load_afrag(Ab + (wm * 32 + 16 + r) * KSTR + 32 * s, hlf);
      v16bf b0 = load_bfrag(Bb + (wn * 32 + r) * KSTR + 32 * s, hlf);
      v16bf b1 = load_bfrag(Bb + (wn * 32 + 16 + r) * KSTR + 32 * s, hlf);
      acc[0][0] = wmma_bf16(a0, b0, acc[0][0]);
      acc[0][1] = wmma_bf16(a0, b1, acc[0][1]);
      acc[1][0] = wmma_bf16(a1, b0, acc[1][0]);
      acc[1][1] = wmma_bf16(a1, b1, acc[1][1]);
    }
    cur ^= 1;
  }

#pragma unroll
  for (int mi = 0; mi < 2; mi++)
#pragma unroll
    for (int ni = 0; ni < 2; ni++)
#pragma unroll
      for (int v = 0; v < 8; v++) {
        int m = m0 + wm * 32 + mi * 16 + v + 8 * hlf;
        int n = n0 + wn * 32 + ni * 16 + r;
        C[(size_t)m * N + n] = acc[mi][ni][v];
      }
}

// ---------------------------------------------------------------------------
// RoPE + relayout to (B, H, S, 128) bf16. One thread per (b,s,h,d/2) pair.
// Input: xin f32 (B*S, H*128) row-major.
// ---------------------------------------------------------------------------
__global__ void rope_to_bf16(const float* __restrict__ xin,
                             const float* __restrict__ cosb,
                             const float* __restrict__ sinb,
                             __bf16* __restrict__ outp,
                             int H, size_t n) {
  size_t i = (size_t)blockIdx.x * 256u + threadIdx.x;
  if (i >= n) return;
  int d2   = (int)(i & 63);
  size_t t = i >> 6;
  int h    = (int)(t % H);
  size_t bs = t / H;           // b*SEQ + s
  int s    = (int)(bs % SEQ);
  size_t b = bs / SEQ;

  size_t src = (bs * H + h) * HD + 2 * d2;
  float x0 = xin[src], x1 = xin[src + 1];
  float c = cosb[(size_t)s * 64 + d2], sn = sinb[(size_t)s * 64 + d2];
  float o0 = x0 * c - x1 * sn;
  float o1 = x0 * sn + x1 * c;
  size_t dst = (((b * H + h) * (size_t)SEQ + s) * HD) + 2 * d2;
  outp[dst]     = (__bf16)o0;
  outp[dst + 1] = (__bf16)o1;
}

// V: convert + relayout to (B, Hkv, S, 128) bf16, no rotation.
__global__ void v_to_bf16(const float* __restrict__ xin, __bf16* __restrict__ outp,
                          size_t n) {
  size_t i = (size_t)blockIdx.x * 256u + threadIdx.x;
  if (i >= n) return;
  int d    = (int)(i & 127);
  size_t t = i >> 7;
  int h    = (int)(t % N_KV_HEADS);
  size_t bs = t / N_KV_HEADS;
  int s    = (int)(bs % SEQ);
  size_t b = bs / SEQ;
  outp[(((b * N_KV_HEADS + h) * (size_t)SEQ + s) * HD) + d] =
      (__bf16)xin[(bs * N_KV_HEADS + h) * HD + d];
}

// ---------------------------------------------------------------------------
// Flash attention (causal, GQA rep=4).
// Block = (b, h, 128 q-rows), 256 threads = 8 waves; wave w owns a 16-row
// Q tile held as 4 resident A-fragments. K (async) / transposed-V tiles of
// 32 keys staged cooperatively in LDS; online softmax with 16-lane shuffles;
// P bounced through wave-private LDS (C -> A layout) for the PV WMMAs.
// Output: bf16 attn, layout (B*S, N_HEADS*128) for the final projection GEMM.
// ---------------------------------------------------------------------------
#define PSTR 40

__global__ __launch_bounds__(256) void attn_fwd(
    const __bf16* __restrict__ Q, const __bf16* __restrict__ Kc,
    const __bf16* __restrict__ Vc, __bf16* __restrict__ O, float scale) {
  __shared__ __align__(16) __bf16 Ks[32 * 136];       // 32 keys x 128 dims (+pad)
  __shared__ __align__(16) __bf16 Vt[HD * PSTR];      // V^T: 128 dims x 32 keys
  __shared__ __align__(16) __bf16 Pw[8 * 16 * PSTR];  // per-wave P: 16 x 32

  const int b  = blockIdx.z;
  const int h  = blockIdx.y;
  const int q0 = blockIdx.x * 128;
  const int tid = threadIdx.x, lane = tid & 31, w = tid >> 5;
  const int r = lane & 15, hlf = lane >> 4;
  const int kvh = h >> 2;

  const __bf16* Qb = Q  + ((size_t)b * N_HEADS    + h)   * SEQ * HD;
  const __bf16* Kb = Kc + ((size_t)b * N_KV_HEADS + kvh) * SEQ * HD;
  const __bf16* Vb = Vc + ((size_t)b * N_KV_HEADS + kvh) * SEQ * HD;

  // Resident Q fragments: 4 chunks of K=32 covering head_dim 128.
  const int qrow = q0 + w * 16 + r;
  v16bf qf[4];
#pragma unroll
  for (int ch = 0; ch < 4; ch++)
    qf[ch] = load_afrag(Qb + (size_t)qrow * HD + ch * 32, hlf);

  float mrow[8], lrow[8];
  v8f acc[8] = {};
#pragma unroll
  for (int v = 0; v < 8; v++) { mrow[v] = -1e30f; lrow[v] = 0.f; }

  const int srow = tid >> 3;        // 0..31 (key within tile)
  const int scol = (tid & 7) * 16;  // 0..112 (dim)

  const int ktiles = (q0 >> 5) + 4; // causal: keys up to q0+127

  for (int kt = 0; kt < ktiles; kt++) {
    const int kb = kt * 32;
    __syncthreads();
    { // stage K tile (row-major) via async global->LDS
      const __bf16* g = Kb + (size_t)(kb + srow) * HD + scol;
      cp16_g2l(g,     &Ks[srow * 136 + scol]);
      cp16_g2l(g + 8, &Ks[srow * 136 + scol + 8]);
    }
    { // stage V tile transposed: Vt[d][k]
      const __bf16* g = Vb + (size_t)(kb + srow) * HD + scol;
      v8bf v0 = *(const v8bf*)g;
      v8bf v1 = *(const v8bf*)(g + 8);
#pragma unroll
      for (int j = 0; j < 8; j++) {
        Vt[(scol + j) * PSTR + srow]     = v0[j];
        Vt[(scol + 8 + j) * PSTR + srow] = v1[j];
      }
    }
    wait_async();
    __syncthreads();

    // scores: 16 q-rows x 32 keys = two 16x16 C tiles, K-depth 128
    v8f sc0 = {}, sc1 = {};
#pragma unroll
    for (int ch = 0; ch < 4; ch++) {
      v16bf b0 = load_bfrag(&Ks[r * 136 + ch * 32], hlf);
      v16bf b1 = load_bfrag(&Ks[(16 + r) * 136 + ch * 32], hlf);
      sc0 = wmma_bf16(qf[ch], b0, sc0);
      sc1 = wmma_bf16(qf[ch], b1, sc1);
    }

    // online softmax, row = lane-half of one accumulator VGPR
#pragma unroll
    for (int v = 0; v < 8; v++) {
      int qm = q0 + w * 16 + v + 8 * hlf;
      float s0 = sc0[v] * scale + ((kb + r)      > qm ? -1e9f : 0.f);
      float s1 = sc1[v] * scale + ((kb + 16 + r) > qm ? -1e9f : 0.f);
      float rm = fmaxf(s0, s1);
      rm = fmaxf(rm, __shfl_xor(rm, 1, 32));
      rm = fmaxf(rm, __shfl_xor(rm, 2, 32));
      rm = fmaxf(rm, __shfl_xor(rm, 4, 32));
      rm = fmaxf(rm, __shfl_xor(rm, 8, 32));
      float mnew  = fmaxf(mrow[v], rm);
      float alpha = __expf(mrow[v] - mnew);
      float p0 = __expf(s0 - mnew);
      float p1 = __expf(s1 - mnew);
      float rs = p0 + p1;
      rs += __shfl_xor(rs, 1, 32);
      rs += __shfl_xor(rs, 2, 32);
      rs += __shfl_xor(rs, 4, 32);
      rs += __shfl_xor(rs, 8, 32);
      lrow[v] = lrow[v] * alpha + rs;
      mrow[v] = mnew;
#pragma unroll
      for (int dt = 0; dt < 8; dt++) acc[dt][v] *= alpha;
      int m = v + 8 * hlf;  // bounce P through wave-private LDS (C -> A layout)
      Pw[(w * 16 + m) * PSTR + r]      = (__bf16)p0;
      Pw[(w * 16 + m) * PSTR + 16 + r] = (__bf16)p1;
    }
    // same-wave LDS ordering: drain DScnt before re-reading P
    asm volatile("s_wait_dscnt 0" ::: "memory");

    v16bf pf = load_afrag(&Pw[(w * 16 + r) * PSTR], hlf);
#pragma unroll
    for (int dt = 0; dt < 8; dt++) {
      v16bf vf = load_bfrag(&Vt[(dt * 16 + r) * PSTR], hlf);
      acc[dt] = wmma_bf16(pf, vf, acc[dt]);
    }
  }

  // epilogue: O[b*S+q, h*128+d] = acc / l   (bf16 for the final GEMM)
#pragma unroll
  for (int v = 0; v < 8; v++) {
    float inv = 1.0f / lrow[v];
    int qm = q0 + w * 16 + v + 8 * hlf;
    size_t rowb = ((size_t)b * SEQ + qm) * DMODEL + (size_t)h * HD;
#pragma unroll
    for (int dt = 0; dt < 8; dt++)
      O[rowb + dt * 16 + r] = (__bf16)(acc[dt][v] * inv);
  }
}

// ---------------------------------------------------------------------------
// Host launcher
// ---------------------------------------------------------------------------
extern "C" void kernel_launch(void* const* d_in, const int* in_sizes, int n_in,
                              void* d_out, int out_size, void* d_ws, size_t ws_size,
                              hipStream_t stream) {
  const float* x  = (const float*)d_in[0];
  const float* fc = (const float*)d_in[2];
  const float* fs = (const float*)d_in[3];
  const float* wq = (const float*)d_in[7];
  const float* wk = (const float*)d_in[8];
  const float* wv = (const float*)d_in[9];
  const float* wo = (const float*)d_in[10];
  float* out = (float*)d_out;

  const size_t MR = (size_t)BATCH * SEQ;  // 4096 rows

  char* wsp = (char*)d_ws;
  auto alloc = [&](size_t bytes) -> char* {
    char* p = wsp;
    wsp += (bytes + 255) & ~(size_t)255;
    return p;
  };
  __bf16* xbf   = (__bf16*)alloc(MR * DMODEL * 2);
  __bf16* wqbf  = (__bf16*)alloc((size_t)DMODEL * DMODEL * 2);
  __bf16* wkbf  = (__bf16*)alloc((size_t)KVD * DMODEL * 2);
  __bf16* wvbf  = (__bf16*)alloc((size_t)KVD * DMODEL * 2);
  __bf16* wobf  = (__bf16*)alloc((size_t)DMODEL * DMODEL * 2);
  float*  xqf   = (float*)alloc(MR * DMODEL * 4);
  float*  xkf   = (float*)alloc(MR * KVD * 4);
  float*  xvf   = (float*)alloc(MR * KVD * 4);
  __bf16* Qb    = (__bf16*)alloc(MR * DMODEL * 2);
  __bf16* Kb    = (__bf16*)alloc(MR * KVD * 2);
  __bf16* Vb    = (__bf16*)alloc(MR * KVD * 2);
  __bf16* attnb = (__bf16*)xqf;  // alias: xq f32 is dead after RoPE-Q

  auto cvt = [&](const float* src, __bf16* dst, size_t n) {
    cvt_f32_bf16<<<dim3((unsigned)((n + 255) / 256)), 256, 0, stream>>>(src, dst, n);
  };
  // 1) precision conversion
  cvt(x,  xbf,  MR * DMODEL);
  cvt(wq, wqbf, (size_t)DMODEL * DMODEL);
  cvt(wk, wkbf, (size_t)KVD * DMODEL);
  cvt(wv, wvbf, (size_t)KVD * DMODEL);
  cvt(wo, wobf, (size_t)DMODEL * DMODEL);

  // 2) QKV projections (NT GEMM)
  gemm_nt_bf16<<<dim3(DMODEL / BN, MR / BM), 256, 0, stream>>>(
      xbf, wqbf, xqf, (int)MR, DMODEL, DMODEL);
  gemm_nt_bf16<<<dim3(KVD / BN, MR / BM), 256, 0, stream>>>(
      xbf, wkbf, xkf, (int)MR, KVD, DMODEL);
  gemm_nt_bf16<<<dim3(KVD / BN, MR / BM), 256, 0, stream>>>(
      xbf, wvbf, xvf, (int)MR, KVD, DMODEL);

  // 3) RoPE + relayout to (B,H,S,128) bf16
  {
    size_t n = MR * N_HEADS * 64;
    rope_to_bf16<<<dim3((unsigned)((n + 255) / 256)), 256, 0, stream>>>(
        xqf, fc, fs, Qb, N_HEADS, n);
  }
  {
    size_t n = MR * N_KV_HEADS * 64;
    rope_to_bf16<<<dim3((unsigned)((n + 255) / 256)), 256, 0, stream>>>(
        xkf, fc, fs, Kb, N_KV_HEADS, n);
  }
  {
    size_t n = MR * N_KV_HEADS * 128;
    v_to_bf16<<<dim3((unsigned)((n + 255) / 256)), 256, 0, stream>>>(xvf, Vb, n);
  }

  // 4) causal flash attention
  attn_fwd<<<dim3(SEQ / 128, N_HEADS, BATCH), 256, 0, stream>>>(
      Qb, Kb, Vb, attnb, 1.0f / sqrtf((float)HD));

  // 5) output projection -> f32 d_out
  gemm_nt_bf16<<<dim3(DMODEL / BN, MR / BM), 256, 0, stream>>>(
      attnb, wobf, out, (int)MR, DMODEL, DMODEL);
}
